// Encoder_60816736911763
// MI455X (gfx1250) — matrix operand order repaired and checked by reading, not compiled, not run on hardware
//
#include <hip/hip_runtime.h>
#include <hip/hip_bf16.h>
#include <math.h>
#include <stdint.h>

// ---------------------------------------------------------------------------
// 2-layer tanh RNN encoder for MI455X (gfx1250, wave32, WMMA).
//   out[l] = final hidden state of layer l, shape [B, HID], fp32.
// Strategy:
//   * All matmuls in bf16 WMMA (v_wmma_f32_16x16x32_bf16), fp32 accumulate.
//   * embed converted once to bf16; layer-0 GEMM gathers bf16 rows directly.
//   * GEMM: 128x128 tiles, double-buffered LDS staged with
//     GLOBAL_LOAD_ASYNC_TO_LDS_B128 (ASYNCcnt path) when available.
//   * xb stored transposed [t][n][b] so the latency-critical scan reads one
//     b128 per lane per step instead of 8 scattered u16.
//   * Scan: 16 WGs, W_hh slice resident in LDS, h ping-pong in L2, one
//     atomic grid barrier per timestep.
// ---------------------------------------------------------------------------

typedef __attribute__((ext_vector_type(16))) __bf16   v16bf;
typedef __attribute__((ext_vector_type(8)))  float    v8f;
typedef __attribute__((ext_vector_type(4)))  unsigned v4u;
typedef __attribute__((ext_vector_type(4)))  float    v4f;

// Pointee/pointer types matching the async-LDS builtin's exact signature:
// (int __vector(4) addrspace(1)*, int __vector(4) addrspace(3)*, Ii, Ii).
typedef int v4i_vs __attribute__((vector_size(16)));
typedef __attribute__((address_space(1))) v4i_vs* gptr_v4i;
typedef __attribute__((address_space(3))) v4i_vs* lptr_v4i;

union FragAB { v4u u[2]; v16bf v; };   // 32B: one 16x32 bf16 A/B fragment

static constexpr int Bk = 64;      // batch
static constexpr int Sk = 2048;    // sequence length
static constexpr int Dk = 512;     // input dim
static constexpr int Hk = 512;     // hidden dim
static constexpr int Vk = 32000;   // vocab
static constexpr int NWG_SCAN = 16;

#if defined(__has_builtin)
#if __has_builtin(__builtin_amdgcn_global_load_async_to_lds_b128)
#define ASYNC_LDS 1
#endif
#endif

__device__ __forceinline__ unsigned short f2bf(float f) {
  union { float f; unsigned u; } v; v.f = f;
  unsigned r = v.u + 0x7FFFu + ((v.u >> 16) & 1u);   // round-to-nearest-even
  return (unsigned short)(r >> 16);
}
__device__ __forceinline__ float bf2f(unsigned short h) {
  union { unsigned u; float f; } v; v.u = ((unsigned)h) << 16; return v.f;
}

// Global->LDS 16B copy: async (no VGPR round-trip, ASYNCcnt) when available.
// Generic LDS pointers carry the LDS byte offset in their low 32 bits, so an
// integer-route cast to AS(3) is valid; global generic == AS(1) bit pattern.
__device__ __forceinline__ void g2l_b128(const void* g, void* l) {
#ifdef ASYNC_LDS
  __builtin_amdgcn_global_load_async_to_lds_b128(
      (gptr_v4i)(uintptr_t)g, (lptr_v4i)(unsigned)(uintptr_t)l, 0, 0);
#else
  *(v4u*)l = *(const v4u*)g;
#endif
}
__device__ __forceinline__ void g2l_wait() {
#ifdef ASYNC_LDS
#if __has_builtin(__builtin_amdgcn_s_wait_asynccnt)
  __builtin_amdgcn_s_wait_asynccnt(0);
#else
  asm volatile("s_wait_asynccnt 0" ::: "memory");
#endif
#endif
}

// ---------------------------------------------------------------------------
// One-time packs: weights fp32->bf16 (layout [layer][n][k] = B-fragment
// friendly), folded bias, and the bf16 embedding table.
// ---------------------------------------------------------------------------
__global__ void pack_weights(const float* __restrict__ Wih,
                             const float* __restrict__ Whh,
                             const float* __restrict__ bih,
                             const float* __restrict__ bhh,
                             unsigned short* __restrict__ WihB,
                             unsigned short* __restrict__ WhhB,
                             float* __restrict__ bias) {
  int i = blockIdx.x * blockDim.x + threadIdx.x;
  if (i < 2 * Hk * Dk) {
    WihB[i] = f2bf(Wih[i]);
    WhhB[i] = f2bf(Whh[i]);
  }
  if (i < 2 * Hk) bias[i] = bih[i] + bhh[i];
}

__global__ void pack_embed(const float* __restrict__ E,
                           unsigned short* __restrict__ Eb, int n8) {
  int i = blockIdx.x * blockDim.x + threadIdx.x;
  if (i >= n8) return;
  v4f f0 = *(const v4f*)(E + (size_t)i * 8);
  v4f f1 = *(const v4f*)(E + (size_t)i * 8 + 4);
  union { unsigned short h[8]; v4u u; } pk;
  pk.h[0] = f2bf(f0.x); pk.h[1] = f2bf(f0.y);
  pk.h[2] = f2bf(f0.z); pk.h[3] = f2bf(f0.w);
  pk.h[4] = f2bf(f1.x); pk.h[5] = f2bf(f1.y);
  pk.h[6] = f2bf(f1.z); pk.h[7] = f2bf(f1.w);
  *(v4u*)(Eb + (size_t)i * 8) = pk.u;
}

// Zero the h ping-pong buffers (h0 = 0) and grid-barrier counter, every
// launch (graph-replay deterministic).
__global__ void init_scan(unsigned* __restrict__ hbuf_u,
                          unsigned* __restrict__ syncc) {
  int i = blockIdx.x * blockDim.x + threadIdx.x;
  if (i < 2 * Bk * Hk / 2) hbuf_u[i] = 0;
  if (i == 0) *syncc = 0;
}

// ---------------------------------------------------------------------------
// xbT[t][n][b] = (A @ W^T + bias), A = gathered bf16 embed rows (layer 0) or
// layer-0 outputs (bf16 [M][Dk]).  M = Sk*Bk, N = K = 512.
// 256 thr / 8 waves, 128x128 tile, K staged 32 at a time through
// double-buffered LDS (rows padded to 40 halfwords -> conflict-free b128
// fragment reads).  Async staging overlaps stage k+1 with stage-k WMMAs.
// ---------------------------------------------------------------------------
__global__ __launch_bounds__(256)
void gemm_xb(const unsigned short* __restrict__ Abf,   // bf16 [M][Dk] or null
             const unsigned short* __restrict__ embB,  // bf16 [Vk][Dk]
             const int*   __restrict__ src,            // [Bk][Sk]
             const unsigned short* __restrict__ Wb,    // bf16 [Hk][Dk] layer
             const float* __restrict__ bias,           // [Hk] layer
             unsigned short* __restrict__ xbT,         // bf16 [Sk][Hk][Bk]
             int gather) {
  __shared__ alignas(16) unsigned short As[2][128][40];
  __shared__ alignas(16) unsigned short Bs[2][128][40];

  const int tid  = threadIdx.x;
  const int m0   = blockIdx.x * 128;
  const int n0   = blockIdx.y * 128;
  const int w    = tid >> 5;
  const int lane = tid & 31;
  const int lo   = lane & 15;
  const int hi   = lane >> 4;
  const int wm   = w & 3;        // 4 waves along M (4*32 = 128)
  const int wn   = w >> 2;       // 2 waves along N (2*64 = 128)

  // Per-thread staging slots: 2 b128 units each for A and B per stage.
  const unsigned short* gA[2];
  const unsigned short* gB[2];
  unsigned short* lA[2][2];
  unsigned short* lB[2][2];
  for (int i = 0; i < 2; ++i) {
    const int f = tid * 2 + i, row = f >> 2, q = f & 3;
    const int m = m0 + row;
    if (gather) {
      const int s = m >> 6, b = m & 63;          // m = s*Bk + b
      gA[i] = embB + (size_t)src[(size_t)b * Sk + s] * Dk + q * 8;
    } else {
      gA[i] = Abf + (size_t)m * Dk + q * 8;
    }
    gB[i] = Wb + (size_t)(n0 + row) * Dk + q * 8;
    for (int u = 0; u < 2; ++u) {
      lA[u][i] = &As[u][row][q * 8];
      lB[u][i] = &Bs[u][row][q * 8];
    }
  }
  auto stage = [&](int kk) {
    const int u = kk & 1, ko = kk * 32;
    for (int i = 0; i < 2; ++i) {
      g2l_b128(gA[i] + ko, lA[u][i]);
      g2l_b128(gB[i] + ko, lB[u][i]);
    }
  };

  v8f acc[2][4] = {};            // 2 M-tiles x 4 N-tiles of 16x16 fp32
  stage(0);
  for (int kk = 0; kk < Dk / 32; ++kk) {
    g2l_wait();                  // my stage-kk async writes done
    __syncthreads();             // everyone's writes visible, prior reads done
    if (kk + 1 < Dk / 32) stage(kk + 1);   // overlap with compute
    const int u = kk & 1;

    FragAB a[2], b[4];
    for (int tm = 0; tm < 2; ++tm) {
      const int r  = wm * 32 + tm * 16 + lo;
      const int kb = hi * 8;                     // lanes 0-15: K{0-7,16-23}
      a[tm].u[0] = *(const v4u*)&As[u][r][kb];
      a[tm].u[1] = *(const v4u*)&As[u][r][kb + 16];
    }
    for (int tn = 0; tn < 4; ++tn) {
      const int r  = wn * 64 + tn * 16 + lo;
      const int kb = hi * 8;                     // lanes 0-15: K{0-7,16-23}
      b[tn].u[0] = *(const v4u*)&Bs[u][r][kb];
      b[tn].u[1] = *(const v4u*)&Bs[u][r][kb + 16];
    }
    for (int tm = 0; tm < 2; ++tm)
      for (int tn = 0; tn < 4; ++tn)
        acc[tm][tn] = __builtin_amdgcn_wmma_f32_16x16x32_bf16(
            false, a[tm].v, false, b[tn].v, (short)0, acc[tm][tn],
            false, false);
  }

  // Epilogue: + bias, fp32->bf16, one b128 store per tile per lane into the
  // transposed layout (lane owns fixed n, 8 consecutive b within one t).
  for (int tn = 0; tn < 4; ++tn) {
    const int   n_e  = n0 + wn * 64 + tn * 16 + lo;
    const float bsum = bias[n_e];
    for (int tm = 0; tm < 2; ++tm) {
      const int mb = m0 + wm * 32 + tm * 16 + hi * 8;   // 8-aligned
      const int t = mb >> 6, b0 = mb & 63;
      union { unsigned short h[8]; v4u u; } pk;
      for (int v = 0; v < 8; ++v) pk.h[v] = f2bf(acc[tm][tn][v] + bsum);
      *(v4u*)(xbT + ((size_t)t * Hk + n_e) * Bk + b0) = pk.u;
    }
  }
}

// ---------------------------------------------------------------------------
// Recurrent scan: h_{t+1} = tanh(xb[t] + h_t @ Whh^T).
// 16 WGs; WG g owns n-slice [g*32, g*32+32) with its Whh slice in LDS.
// h ping-pongs in global (L2 resident).  One atomic barrier per step.
// Two accumulators break the 16-deep WMMA dependency chain.
// ---------------------------------------------------------------------------
__global__ __launch_bounds__(256)
void rnn_scan(const unsigned short* __restrict__ WhhB,  // bf16 [Hk][Hk] layer
              const unsigned short* __restrict__ xbT,   // bf16 [Sk][Hk][Bk]
              unsigned short* __restrict__ ybuf,        // bf16 [Sk*Bk][Hk]
              unsigned short* __restrict__ hbuf,        // bf16 [2][Bk][Hk]
              float* __restrict__ outh,                 // fp32 [Bk][Hk]
              unsigned* __restrict__ syncc) {
  __shared__ alignas(16) unsigned short Wsl[32][520];   // 520*2B = 65*16B

  const int tid = threadIdx.x;
  const int ns  = blockIdx.x * 32;                      // n-slice base

  for (int i = 0; i < 8; ++i) {                         // 32x512 bf16 slice
    const int f = i * 256 + tid, row = f >> 6, q = f & 63;
    *(v4u*)&Wsl[row][q * 8] =
        *(const v4u*)(WhhB + (size_t)(ns + row) * Hk + q * 8);
  }
  __syncthreads();

  const int w    = tid >> 5;
  const int lane = tid & 31;
  const int lo   = lane & 15;
  const int hi   = lane >> 4;
  const int tm   = w & 3;             // 4 waves along batch (4*16 = 64)
  const int tn   = w >> 2;            // 2 waves along the 32-wide n slice
  const int mbase = tm * 16;
  const int nloc  = tn * 16 + lo;
  const int n_e   = ns + nloc;

  unsigned target = 0;
  for (int t = 0; t < Sk; ++t) {
    const unsigned short* hcur = hbuf + (size_t)(t & 1) * Bk * Hk;
    unsigned short*       hnxt = hbuf + (size_t)((t & 1) ^ 1) * Bk * Hk;

    // Independent of the matmul: fetch this lane's xb vector early.
    union { v4u u; unsigned short h[8]; } xv;
    xv.u = *(const v4u*)(xbT + ((size_t)t * Hk + n_e) * Bk + mbase + hi * 8);

    v8f acc0 = {}, acc1 = {};
    for (int kk = 0; kk < Hk / 32; kk += 2) {
      FragAB a0, b0, a1, b1;
      const int k0 = kk * 32, k1 = k0 + 32;
      const unsigned short* hp = hcur + (size_t)(mbase + lo) * Hk + hi * 8;
      a0.u[0] = *(const v4u*)(hp + k0);
      a0.u[1] = *(const v4u*)(hp + k0 + 16);
      a1.u[0] = *(const v4u*)(hp + k1);
      a1.u[1] = *(const v4u*)(hp + k1 + 16);
      b0.u[0] = *(const v4u*)&Wsl[nloc][k0 + hi * 8];
      b0.u[1] = *(const v4u*)&Wsl[nloc][k0 + hi * 8 + 16];
      b1.u[0] = *(const v4u*)&Wsl[nloc][k1 + hi * 8];
      b1.u[1] = *(const v4u*)&Wsl[nloc][k1 + hi * 8 + 16];
      acc0 = __builtin_amdgcn_wmma_f32_16x16x32_bf16(
          false, a0.v, false, b0.v, (short)0, acc0, false, false);
      acc1 = __builtin_amdgcn_wmma_f32_16x16x32_bf16(
          false, a1.v, false, b1.v, (short)0, acc1, false, false);
    }
    const v8f acc = acc0 + acc1;

    const size_t xoff = (size_t)t * Bk * Hk;
    for (int v = 0; v < 8; ++v) {
      const int    m  = mbase + hi * 8 + v;
      const size_t o  = (size_t)m * Hk + n_e;
      const float  hv = tanhf(acc[v] + bf2f(xv.h[v]));
      const unsigned short hb = f2bf(hv);
      hnxt[o]        = hb;             // next-step A matrix (row-major in k)
      ybuf[xoff + o] = hb;             // next layer's GEMM input
      if (t == Sk - 1) outh[o] = hv;
    }

    // Grid barrier (monotonic counter, zeroed by init_scan each launch).
    target += NWG_SCAN;
    __threadfence();
    __syncthreads();
    if (tid == 0) {
      atomicAdd(syncc, 1u);
      while (__hip_atomic_load(syncc, __ATOMIC_ACQUIRE,
                               __HIP_MEMORY_SCOPE_AGENT) < target) {
        __builtin_amdgcn_s_sleep(1);
      }
    }
    __syncthreads();
    __threadfence();
  }
}

// ---------------------------------------------------------------------------
extern "C" void kernel_launch(void* const* d_in, const int* in_sizes, int n_in,
                              void* d_out, int out_size, void* d_ws,
                              size_t ws_size, hipStream_t stream) {
  const int*   src   = (const int*)d_in[0];
  const float* embed = (const float*)d_in[1];
  const float* Wih   = (const float*)d_in[2];
  const float* Whh   = (const float*)d_in[3];
  const float* bih   = (const float*)d_in[4];
  const float* bhh   = (const float*)d_in[5];
  float* out = (float*)d_out;

  char* ws = (char*)d_ws;
  const size_t SBH = (size_t)Sk * Bk * Hk;
  unsigned short* xbT  = (unsigned short*)ws; ws += SBH * 2;            // 134MB
  unsigned short* yB   = (unsigned short*)ws; ws += SBH * 2;            // 134MB
  unsigned short* embB = (unsigned short*)ws; ws += (size_t)Vk * Dk * 2;// 33MB
  unsigned short* WihB = (unsigned short*)ws; ws += (size_t)2 * Hk * Dk * 2;
  unsigned short* WhhB = (unsigned short*)ws; ws += (size_t)2 * Hk * Hk * 2;
  float*          bias = (float*)ws;          ws += (size_t)2 * Hk * 4;
  unsigned short* hbuf = (unsigned short*)ws; ws += (size_t)2 * Bk * Hk * 2;
  unsigned*       sync = (unsigned*)ws;

  pack_weights<<<(2 * Hk * Dk) / 256, 256, 0, stream>>>(
      Wih, Whh, bih, bhh, WihB, WhhB, bias);
  pack_embed<<<(Vk * Dk / 8 + 255) / 256, 256, 0, stream>>>(
      embed, embB, Vk * Dk / 8);

  // ---- layer 0: gather-fused input GEMM, then scan ------------------------
  init_scan<<<(2 * Bk * Hk / 2 + 255) / 256, 256, 0, stream>>>(
      (unsigned*)hbuf, sync);
  gemm_xb<<<dim3((Sk * Bk) / 128, Hk / 128), 256, 0, stream>>>(
      nullptr, embB, src, WihB, bias, xbT, /*gather=*/1);
  rnn_scan<<<NWG_SCAN, 256, 0, stream>>>(WhhB, xbT, yB, hbuf, out, sync);

  // ---- layer 1: input GEMM on layer-0 outputs, then scan ------------------
  init_scan<<<(2 * Bk * Hk / 2 + 255) / 256, 256, 0, stream>>>(
      (unsigned*)hbuf, sync);
  gemm_xb<<<dim3((Sk * Bk) / 128, Hk / 128), 256, 0, stream>>>(
      yB, embB, src, WihB + (size_t)Hk * Dk, bias + Hk, xbT, /*gather=*/0);
  rnn_scan<<<NWG_SCAN, 256, 0, stream>>>(WhhB + (size_t)Hk * Hk, xbT, yB, hbuf,
                                         out + (size_t)Bk * Hk, sync);
}
